// DenseMoE_79336635892596
// MI455X (gfx1250) — compile-verified, matrix operand order checked by reading, not run on hardware
//
#include <hip/hip_runtime.h>
#include <hip/hip_bf16.h>

// ---------------------------------------------------------------------------
// DeepSeek-style dense MoE for MI455X (gfx1250, wave32, WMMA bf16 path)
//   x:[T,H] f32, Wg:[E,H], Wgu:[E,2I,H], Wd:[E,H,I], Wgu_s:[2*Is,H], Wd_s:[H,Is]
//   T=1024 H=2048 I=1408 E=32 Is=2816, top-k=6 of 32 with 8 groups/top-3 groups
//
// Pipeline:
//   1) x -> bf16 (xb)
//   2) router -> dense combine weights w[T,E]
//   3) per expert: H[e] = w[:,e] * swiglu(x @ Wgu[e]^T)      (weight folded in!)
//      shared:     Hs  =          swiglu(x @ Wgu_s^T)
//   4) out  = concat_e(H[e]) @ concat_e(Wd[e])^T             (one GEMM, split-K)
//      out += Hs @ Wd_s^T                                     (same kernel)
// ---------------------------------------------------------------------------

#define T_TOK   1024
#define HID     2048
#define I_MOE   1408
#define N_EXP   32
#define N_GROUP 8
#define TOPKG   3
#define TOP_K   6
#define I_SH    2816

typedef __attribute__((ext_vector_type(16))) __bf16 v16bf;
typedef __attribute__((ext_vector_type(8)))  __bf16 v8bf;
typedef __attribute__((ext_vector_type(8)))  float  v8f;

// pack two f32 -> packed bf16 dword (round-half-up; lo in [15:0], hi in [31:16])
__device__ __forceinline__ unsigned pk2bf(float lo, float hi) {
  unsigned ul = __builtin_bit_cast(unsigned, lo) + 0x8000u;
  unsigned uh = __builtin_bit_cast(unsigned, hi) + 0x8000u;
  return (uh & 0xFFFF0000u) | (ul >> 16);
}

__device__ __forceinline__ __bf16 f2bf(float f) {
  unsigned u = (__builtin_bit_cast(unsigned, f) + 0x8000u) >> 16;
  unsigned short s = (unsigned short)u;
  return __builtin_bit_cast(__bf16, s);
}

__device__ __forceinline__ v8f vzero8() {
  v8f z; for (int i = 0; i < 8; ++i) z[i] = 0.0f; return z;
}

// Load one 16x32 bf16 A/B fragment from LDS (row-major, ld in elements).
// ISA layout (CDNA5 ISA): lanes 0-15 row M=lane hold K octets {0..7,16..23},
// lanes 16-31 hold {8..15,24..31}. Two ds_load_b128 per fragment.
__device__ __forceinline__ v16bf load_frag(const __bf16* base, int lane, int ld) {
  int r    = lane & 15;
  int half = lane >> 4;
  const __bf16* p = base + r * ld + half * 8;
  v16bf f;
  *((v8bf*)&f)       = *((const v8bf*)(p));
  *(((v8bf*)&f) + 1) = *((const v8bf*)(p + 16));
  return f;
}

// async copy 32 bytes global(bf16) -> LDS, tracked by ASYNCcnt
__device__ __forceinline__ void async_cp32(unsigned lds_byte, const __bf16* g) {
  unsigned long ga = (unsigned long)g;
  asm volatile("global_load_async_to_lds_b128 %0, %1, off"
               :: "v"(lds_byte), "v"(ga) : "memory");
  asm volatile("global_load_async_to_lds_b128 %0, %1, off offset:16"
               :: "v"(lds_byte), "v"(ga) : "memory");
}
__device__ __forceinline__ void wait_async0() {
  asm volatile("s_wait_asynccnt 0x0" ::: "memory");
}

// stage one 32-fp32 row chunk as packed bf16 into LDS (four b128 stores)
__device__ __forceinline__ void stage_row32(__bf16* dst, const float* src) {
  float4 a0 = *(const float4*)(src);
  float4 a1 = *(const float4*)(src + 4);
  float4 a2 = *(const float4*)(src + 8);
  float4 a3 = *(const float4*)(src + 12);
  float4 a4 = *(const float4*)(src + 16);
  float4 a5 = *(const float4*)(src + 20);
  float4 a6 = *(const float4*)(src + 24);
  float4 a7 = *(const float4*)(src + 28);
  uint4 w0 = { pk2bf(a0.x, a0.y), pk2bf(a0.z, a0.w),
               pk2bf(a1.x, a1.y), pk2bf(a1.z, a1.w) };
  uint4 w1 = { pk2bf(a2.x, a2.y), pk2bf(a2.z, a2.w),
               pk2bf(a3.x, a3.y), pk2bf(a3.z, a3.w) };
  uint4 w2 = { pk2bf(a4.x, a4.y), pk2bf(a4.z, a4.w),
               pk2bf(a5.x, a5.y), pk2bf(a5.z, a5.w) };
  uint4 w3 = { pk2bf(a6.x, a6.y), pk2bf(a6.z, a6.w),
               pk2bf(a7.x, a7.y), pk2bf(a7.z, a7.w) };
  *(uint4*)(dst)      = w0;
  *(uint4*)(dst + 8)  = w1;
  *(uint4*)(dst + 16) = w2;
  *(uint4*)(dst + 24) = w3;
}

// stage a 16-fp32 half-row as packed bf16 (two b128 stores)
__device__ __forceinline__ void stage_half16(__bf16* dst, const float* src) {
  float4 a0 = *(const float4*)(src);
  float4 a1 = *(const float4*)(src + 4);
  float4 a2 = *(const float4*)(src + 8);
  float4 a3 = *(const float4*)(src + 12);
  uint4 w0 = { pk2bf(a0.x, a0.y), pk2bf(a0.z, a0.w),
               pk2bf(a1.x, a1.y), pk2bf(a1.z, a1.w) };
  uint4 w1 = { pk2bf(a2.x, a2.y), pk2bf(a2.z, a2.w),
               pk2bf(a3.x, a3.y), pk2bf(a3.z, a3.w) };
  *(uint4*)(dst)     = w0;
  *(uint4*)(dst + 8) = w1;
}

// ---------------------------------------------------------------------------
// fp32 -> packed bf16 convert (x -> xb), grid-stride over pairs
// ---------------------------------------------------------------------------
__global__ void k_cvt_bf16(const float* __restrict__ in, unsigned* __restrict__ out,
                           long npair) {
  long i = (long)blockIdx.x * blockDim.x + threadIdx.x;
  long stride = (long)gridDim.x * blockDim.x;
  for (; i < npair; i += stride) {
    float2 v = ((const float2*)in)[i];
    out[i] = pk2bf(v.x, v.y);
  }
}

// ---------------------------------------------------------------------------
// Router: logits + softmax + grouped top-k -> dense combine matrix w[T, E]
// One wave32 per token: lane = expert.
// ---------------------------------------------------------------------------
__global__ void __launch_bounds__(128)
k_router(const float* __restrict__ X, const float* __restrict__ Wg,
         float* __restrict__ wout) {
  int wv   = threadIdx.x >> 5;
  int lane = threadIdx.x & 31;
  int t    = blockIdx.x * 4 + wv;
  if (t >= T_TOK) return;

  const float* x  = X  + (long)t    * HID;
  const float* wr = Wg + (long)lane * HID;
  float acc = 0.0f;
  for (int h = 0; h < HID; h += 4) {
    float4 xv = *(const float4*)(x + h);
    float4 gv = *(const float4*)(wr + h);
    acc = fmaf(xv.x, gv.x, acc);
    acc = fmaf(xv.y, gv.y, acc);
    acc = fmaf(xv.z, gv.z, acc);
    acc = fmaf(xv.w, gv.w, acc);
  }
  float m = acc;
  for (int off = 16; off; off >>= 1) m = fmaxf(m, __shfl_xor(m, off, 32));
  float ex = __expf(acc - m);
  float s = ex;
  for (int off = 16; off; off >>= 1) s += __shfl_xor(s, off, 32);
  float score = ex * __builtin_amdgcn_rcpf(s);

  __shared__ float sc[4][32];
  sc[wv][lane] = score;
  __builtin_amdgcn_s_barrier();

  if (lane == 0) {
    float* S = sc[wv];
    float gs[N_GROUP];
    for (int g = 0; g < N_GROUP; ++g) {
      float gm = S[g * 4];
      for (int j = 1; j < 4; ++j) gm = fmaxf(gm, S[g * 4 + j]);
      gs[g] = gm;
    }
    int gmask = 0;
    for (int it = 0; it < TOPKG; ++it) {
      float best = -1e30f; int bg = 0;
      for (int g = 0; g < N_GROUP; ++g)
        if (!((gmask >> g) & 1) && gs[g] > best) { best = gs[g]; bg = g; }
      gmask |= 1 << bg;
    }
    unsigned emask = 0;
    float wrow[N_EXP];
    for (int e = 0; e < N_EXP; ++e) wrow[e] = 0.0f;
    for (int it = 0; it < TOP_K; ++it) {
      float best = -1e30f; int be = 0;
      for (int e = 0; e < N_EXP; ++e) {
        if (!((gmask >> (e >> 2)) & 1)) continue;
        if ((emask >> e) & 1u) continue;
        if (S[e] > best) { best = S[e]; be = e; }
      }
      emask |= 1u << be;
      wrow[be] = best;
    }
    for (int e = 0; e < N_EXP; ++e) wout[(long)t * N_EXP + e] = wrow[e];
  }
}

// ---------------------------------------------------------------------------
// GEMM1 + SwiGLU (+ fold routing weight):
//   H[e] = w[:,e] * ( silu(X @ Wgate[e]^T) * (X @ Wup[e]^T) )   bf16 out
// Dynamic LDS (bytes): A[0,5120) G[5120,10240) U[10240,15360)
// 64x64 block, 4 waves, each wave 2x2(gate)+2x2(up) WMMA tiles. A staged async.
// ---------------------------------------------------------------------------
#define LDS_SWIGLU (3 * 64 * 40 * 2)
__global__ void __launch_bounds__(128)
k_gemm_swiglu(const __bf16* __restrict__ X, const float* __restrict__ Wgu,
              const float* __restrict__ w, __bf16* __restrict__ H,
              int K, int I, long estrideW, long estrideH) {
  extern __shared__ char smem[];
  __bf16* sA = (__bf16*)(smem);
  __bf16* sG = (__bf16*)(smem + 5120);
  __bf16* sU = (__bf16*)(smem + 10240);

  const int e  = blockIdx.z;
  const int m0 = blockIdx.y * 64;
  const int n0 = blockIdx.x * 64;
  const float* W  = Wgu + (long)e * estrideW;
  __bf16*      Hp = H   + (long)e * estrideH;

  const int tid  = threadIdx.x;
  const int lane = tid & 31;
  const int wave = tid >> 5;
  const int wm   = (wave >> 1) * 32;
  const int wn   = (wave & 1) * 32;

  v8f accG[2][2], accU[2][2];
  for (int i = 0; i < 2; ++i)
    for (int j = 0; j < 2; ++j) { accG[i][j] = vzero8(); accU[i][j] = vzero8(); }

  const int row = tid >> 1;            // 0..63
  const int kq  = (tid & 1) * 16;      // 0 or 16
  const unsigned ldsA = (unsigned)(row * 80 + kq * 2);

  const __bf16* gA = X + (long)(m0 + row) * K + kq;
  const float*  gG = W + (long)(n0 + row) * K + kq;
  const float*  gU = W + (long)(I + n0 + row) * K + kq;

  for (int k0 = 0; k0 < K; k0 += 32) {
    async_cp32(ldsA, gA + k0);                        // A: ASYNCcnt path
    stage_half16(sG + row * 40 + kq, gG + k0);        // G/U: f32 -> bf16
    stage_half16(sU + row * 40 + kq, gU + k0);
    __builtin_prefetch(gG + k0 + 32, 0, 1);
    __builtin_prefetch(gU + k0 + 32, 0, 1);
    wait_async0();
    __syncthreads();

    v16bf a0 = load_frag(sA + wm * 40,        lane, 40);
    v16bf a1 = load_frag(sA + (wm + 16) * 40, lane, 40);
    v16bf g0 = load_frag(sG + wn * 40,        lane, 40);
    v16bf g1 = load_frag(sG + (wn + 16) * 40, lane, 40);
    v16bf u0 = load_frag(sU + wn * 40,        lane, 40);
    v16bf u1 = load_frag(sU + (wn + 16) * 40, lane, 40);

    accG[0][0] = __builtin_amdgcn_wmma_f32_16x16x32_bf16(false, a0, false, g0, (short)0, accG[0][0], false, false);
    accG[0][1] = __builtin_amdgcn_wmma_f32_16x16x32_bf16(false, a0, false, g1, (short)0, accG[0][1], false, false);
    accG[1][0] = __builtin_amdgcn_wmma_f32_16x16x32_bf16(false, a1, false, g0, (short)0, accG[1][0], false, false);
    accG[1][1] = __builtin_amdgcn_wmma_f32_16x16x32_bf16(false, a1, false, g1, (short)0, accG[1][1], false, false);
    accU[0][0] = __builtin_amdgcn_wmma_f32_16x16x32_bf16(false, a0, false, u0, (short)0, accU[0][0], false, false);
    accU[0][1] = __builtin_amdgcn_wmma_f32_16x16x32_bf16(false, a0, false, u1, (short)0, accU[0][1], false, false);
    accU[1][0] = __builtin_amdgcn_wmma_f32_16x16x32_bf16(false, a1, false, u0, (short)0, accU[1][0], false, false);
    accU[1][1] = __builtin_amdgcn_wmma_f32_16x16x32_bf16(false, a1, false, u1, (short)0, accU[1][1], false, false);
    __syncthreads();
  }

  // epilogue: h = w[t,e] * silu(gate) * up.
  // Row weights hoisted: one uniform branch, 16 plain loads, no per-element
  // control flow (round-3 asm showed a branch + cond load per element).
  const int cn    = lane & 15;
  const int rbase = (lane >> 4) * 8;
  float wrow[2][8];
  if (w) {
    #pragma unroll
    for (int tm = 0; tm < 2; ++tm)
      #pragma unroll
      for (int r = 0; r < 8; ++r)
        wrow[tm][r] = w[(long)(m0 + wm + tm * 16 + rbase + r) * N_EXP + e];
  } else {
    #pragma unroll
    for (int tm = 0; tm < 2; ++tm)
      #pragma unroll
      for (int r = 0; r < 8; ++r)
        wrow[tm][r] = 1.0f;
  }
  for (int tm = 0; tm < 2; ++tm)
    for (int tn = 0; tn < 2; ++tn)
      #pragma unroll
      for (int r = 0; r < 8; ++r) {
        int mI = m0 + wm + tm * 16 + rbase + r;
        int nI = n0 + wn + tn * 16 + cn;
        float g = accG[tm][tn][r];
        float u = accU[tm][tn][r];
        float sig = __builtin_amdgcn_rcpf(1.0f + __expf(-g));
        float h = wrow[tm][r] * g * sig * u;
        Hp[(long)mI * I + nI] = f2bf(h);
      }
}

// ---------------------------------------------------------------------------
// Combine GEMM (pure, weights pre-folded):  out += A @ B^T  over chunked K.
//   A: bf16, rows t, per-chunk base csA; B: fp32, rows n, per-chunk base csB.
// 64(M) x 128(N) block, 4 waves, wave tile 32x64 = 2x4 WMMA tiles.
// Split-K across blockIdx.z, combined with global f32 atomics.
// Dynamic LDS (bytes): A[0,5120) B[5120,15360)
// ---------------------------------------------------------------------------
#define LDS_COMBINE (64 * 40 * 2 + 128 * 40 * 2)
__global__ void __launch_bounds__(128)
k_gemm_combine(const __bf16* __restrict__ A, const float* __restrict__ B,
               float* __restrict__ out, int Kiter, int N,
               long rsA, long rsB, long csA, long csB, int chunks_per_z) {
  extern __shared__ char smem[];
  __bf16* sA = (__bf16*)(smem);
  __bf16* sB = (__bf16*)(smem + 5120);

  const int m0 = blockIdx.y * 64;
  const int n0 = blockIdx.x * 128;

  const int tid  = threadIdx.x;
  const int lane = tid & 31;
  const int wave = tid >> 5;
  const int wm   = (wave >> 1) * 32;   // 2 wave-rows of 32 M
  const int wn   = (wave & 1) * 64;    // 2 wave-cols of 64 N

  const int rowA = tid >> 1;           // 0..63
  const int kqA  = (tid & 1) * 16;     // 0 or 16
  const unsigned ldsA = (unsigned)(rowA * 80 + kqA * 2);
  const int rowB = tid;                // 0..127 (one full 32-K row per thread)

  v8f acc[2][4];
  for (int i = 0; i < 2; ++i)
    for (int j = 0; j < 4; ++j) acc[i][j] = vzero8();

  const long c0 = (long)blockIdx.z * chunks_per_z;
  for (int cc = 0; cc < chunks_per_z; ++cc) {
    const long ch = c0 + cc;
    const __bf16* gA = A + ch * csA + (long)(m0 + rowA) * rsA + kqA;
    const float*  gB = B + ch * csB + (long)(n0 + rowB) * rsB;

    for (int k0 = 0; k0 < Kiter; k0 += 32) {
      async_cp32(ldsA, gA + k0);
      stage_row32(sB + rowB * 40, gB + k0);
      __builtin_prefetch(gB + k0 + 32, 0, 1);
      wait_async0();
      __syncthreads();

      v16bf a0 = load_frag(sA + wm * 40,        lane, 40);
      v16bf a1 = load_frag(sA + (wm + 16) * 40, lane, 40);
      v16bf b0 = load_frag(sB + (wn + 0)  * 40, lane, 40);
      v16bf b1 = load_frag(sB + (wn + 16) * 40, lane, 40);
      v16bf b2 = load_frag(sB + (wn + 32) * 40, lane, 40);
      v16bf b3 = load_frag(sB + (wn + 48) * 40, lane, 40);

      acc[0][0] = __builtin_amdgcn_wmma_f32_16x16x32_bf16(false, a0, false, b0, (short)0, acc[0][0], false, false);
      acc[0][1] = __builtin_amdgcn_wmma_f32_16x16x32_bf16(false, a0, false, b1, (short)0, acc[0][1], false, false);
      acc[0][2] = __builtin_amdgcn_wmma_f32_16x16x32_bf16(false, a0, false, b2, (short)0, acc[0][2], false, false);
      acc[0][3] = __builtin_amdgcn_wmma_f32_16x16x32_bf16(false, a0, false, b3, (short)0, acc[0][3], false, false);
      acc[1][0] = __builtin_amdgcn_wmma_f32_16x16x32_bf16(false, a1, false, b0, (short)0, acc[1][0], false, false);
      acc[1][1] = __builtin_amdgcn_wmma_f32_16x16x32_bf16(false, a1, false, b1, (short)0, acc[1][1], false, false);
      acc[1][2] = __builtin_amdgcn_wmma_f32_16x16x32_bf16(false, a1, false, b2, (short)0, acc[1][2], false, false);
      acc[1][3] = __builtin_amdgcn_wmma_f32_16x16x32_bf16(false, a1, false, b3, (short)0, acc[1][3], false, false);
      __syncthreads();
    }
  }

  const int cn    = lane & 15;
  const int rbase = (lane >> 4) * 8;
  for (int tm = 0; tm < 2; ++tm)
    for (int tn = 0; tn < 4; ++tn)
      #pragma unroll
      for (int r = 0; r < 8; ++r) {
        int mI = m0 + wm + tm * 16 + rbase + r;
        int nI = n0 + wn + tn * 16 + cn;
        unsafeAtomicAdd(&out[(long)mI * N + nI], acc[tm][tn][r]);
      }
}

// ---------------------------------------------------------------------------
extern "C" void kernel_launch(void* const* d_in, const int* in_sizes, int n_in,
                              void* d_out, int out_size, void* d_ws, size_t ws_size,
                              hipStream_t stream) {
  (void)in_sizes; (void)n_in; (void)ws_size;
  const float* x     = (const float*)d_in[0];   // [T,H]
  const float* Wg    = (const float*)d_in[1];   // [E,H]
  const float* Wgu   = (const float*)d_in[2];   // [E,2I,H]
  const float* Wd    = (const float*)d_in[3];   // [E,H,I]
  const float* Wgu_s = (const float*)d_in[4];   // [2*Is,H]
  const float* Wd_s  = (const float*)d_in[5];   // [H,Is]
  float* out = (float*)d_out;                   // [T,H]

  char* p = (char*)d_ws;
  __bf16* xb  = (__bf16*)p; p += (size_t)T_TOK * HID * 2;            // 4 MB
  float*  wb  = (float*)p;  p += (size_t)T_TOK * N_EXP * 4;          // 128 KB
  __bf16* Hr  = (__bf16*)p; p += (size_t)N_EXP * T_TOK * I_MOE * 2;  // 88 MB
  __bf16* Hs  = (__bf16*)p;                                          // 5.5 MB

  hipMemsetAsync(out, 0, (size_t)out_size * sizeof(float), stream);

  // 1) x -> bf16 (packed pair writes)
  k_cvt_bf16<<<1024, 256, 0, stream>>>(x, (unsigned*)xb, (long)T_TOK * HID / 2);

  // 2) router -> dense combine weights
  k_router<<<T_TOK / 4, 128, 0, stream>>>(x, Wg, wb);

  // 3) routed experts: gate/up GEMM + SwiGLU + weight fold
  k_gemm_swiglu<<<dim3(I_MOE / 64, T_TOK / 64, N_EXP), 128, LDS_SWIGLU, stream>>>(
      xb, Wgu, wb, Hr, HID, I_MOE, (long)2 * I_MOE * HID, (long)T_TOK * I_MOE);

  // 4) shared expert: gate/up GEMM + SwiGLU (weight = 1)
  k_gemm_swiglu<<<dim3(I_SH / 64, T_TOK / 64, 1), 128, LDS_SWIGLU, stream>>>(
      xb, Wgu_s, nullptr, Hs, HID, I_SH, 0L, 0L);

  // 5) routed combine: one GEMM over K_total = 32*1408, split-K z=4 (8 experts each)
  k_gemm_combine<<<dim3(HID / 128, T_TOK / 64, 4), 128, LDS_COMBINE, stream>>>(
      Hr, Wd, out, I_MOE, HID,
      (long)I_MOE, (long)I_MOE,                     // row strides
      (long)T_TOK * I_MOE, (long)HID * I_MOE,       // chunk (expert) strides
      N_EXP / 4);

  // 6) shared combine: K = 2816 split into 4 chunks of 704 across z
  k_gemm_combine<<<dim3(HID / 128, T_TOK / 64, 4), 128, LDS_COMBINE, stream>>>(
      Hs, Wd_s, out, I_SH / 4, HID,
      (long)I_SH, (long)I_SH,                       // row strides
      (long)(I_SH / 4), (long)(I_SH / 4),           // chunk (k-offset) strides
      1);
}
